// ProteinMPNN_19396072309453
// MI455X (gfx1250) — matrix operand order repaired and checked by reading, not compiled, not run on hardware
//
#include <hip/hip_runtime.h>
#include <hip/hip_bf16.h>
#include <math.h>

typedef __attribute__((ext_vector_type(16))) __bf16 v16bf;
typedef __attribute__((ext_vector_type(8)))  __bf16 v8bf;
typedef __attribute__((ext_vector_type(8)))  float  v8f;

#define BB   2
#define LL   2048
#define KN   48
#define HH   128
#define IN3  384      // 3*H concat input
#define FFD  512      // 4*H FFN hidden
#define NBLK (BB*LL)
#define INV_SCALE (1.0f/30.0f)
#define LN_EPS 1e-5f

union Frag { v16bf v; v8bf h[2]; };

// A-fragment (16x32 bf16, MxK) from an LDS row pointer. Per ISA table:
// lane(hi=0): VGPR0..3 = K 0..7, VGPR4..7 = K 16..23 ; hi=1: +8.
__device__ __forceinline__ v16bf load_a(const __bf16* rowbase, int koff, int hi) {
  Frag f;
  f.h[0] = *(const v8bf*)(rowbase + koff + hi * 8);
  f.h[1] = *(const v8bf*)(rowbase + koff + 16 + hi * 8);
  return f.v;
}

// B-fragment (32x16 bf16, KxN) from pre-transposed weights WT[N][Kdim]:
// lane holds column n = nt*16+lo, K = hi*8..hi*8+7 and 16+hi*8..16+hi*8+7.
__device__ __forceinline__ v16bf load_b(const __bf16* wt, int n, int kdim, int koff, int hi) {
  const __bf16* p = wt + (size_t)n * kdim + koff + hi * 8;
  Frag f;
  f.h[0] = *(const v8bf*)p;
  f.h[1] = *(const v8bf*)(p + 16);
  return f.v;
}

__device__ __forceinline__ v8f wmma_bf16(v16bf a, v16bf b, v8f c) {
  return __builtin_amdgcn_wmma_f32_16x16x32_bf16(false, a, false, b, (short)0, c,
                                                 false, false);
}

__device__ __forceinline__ float gelu_erf(float x) {
  return 0.5f * x * (1.0f + erff(x * 0.70710678118654752440f));
}

// Tree-sum over red[0..127]; must be called by all 256 threads.
__device__ __forceinline__ float blocksum128(float* red, int tid) {
  __syncthreads();
#pragma unroll
  for (int s = 64; s > 0; s >>= 1) {
    if (tid < s) red[tid] += red[tid + s];
    __syncthreads();
  }
  float r = red[0];
  __syncthreads();
  return r;
}

// 3-layer MLP on sA [KN][IN3] (bf16). Layer1->sM (bf16, gelu), layer2->sA-low
// (bf16, gelu, aliased as sM2), layer3 accumulators returned (no epilogue).
// Each of 8 waves owns N-tile w; M tiles 0..2 cover the 48 rows.
__device__ __forceinline__ void mlp3_accum(
    __bf16* sA, __bf16* sM,
    const __bf16* WT1, const float* b1v,
    const __bf16* WT2, const float* b2v,
    const __bf16* WT3, const float* b3v,
    int tid, v8f acc3[3], float* bias3_out) {
  const int w = tid >> 5, lane = tid & 31, hi = lane >> 4, lo = lane & 15;
  const int ncol = w * 16 + lo;

  // ---- layer 1: [48,384] x [384,128] ----
  {
    v8f acc[3] = {};
    for (int kt = 0; kt < IN3 / 32; ++kt) {
      v16bf bf = load_b(WT1, ncol, IN3, kt * 32, hi);
#pragma unroll
      for (int mt = 0; mt < 3; ++mt) {
        v16bf af = load_a(sA + (mt * 16 + lo) * IN3, kt * 32, hi);
        acc[mt] = wmma_bf16(af, bf, acc[mt]);
      }
    }
    float bias = b1v[ncol];
#pragma unroll
    for (int mt = 0; mt < 3; ++mt)
#pragma unroll
      for (int r = 0; r < 8; ++r) {
        int row = mt * 16 + hi * 8 + r;
        sM[row * HH + ncol] = (__bf16)gelu_erf(acc[mt][r] + bias);
      }
  }
  __syncthreads();

  // ---- layer 2: [48,128] x [128,128], sM -> sM2 (= sA low 12288B) ----
  {
    v8f acc[3] = {};
    for (int kt = 0; kt < HH / 32; ++kt) {
      v16bf bf = load_b(WT2, ncol, HH, kt * 32, hi);
#pragma unroll
      for (int mt = 0; mt < 3; ++mt) {
        v16bf af = load_a(sM + (mt * 16 + lo) * HH, kt * 32, hi);
        acc[mt] = wmma_bf16(af, bf, acc[mt]);
      }
    }
    float bias = b2v[ncol];
    __bf16* sM2 = sA;
#pragma unroll
    for (int mt = 0; mt < 3; ++mt)
#pragma unroll
      for (int r = 0; r < 8; ++r) {
        int row = mt * 16 + hi * 8 + r;
        sM2[row * HH + ncol] = (__bf16)gelu_erf(acc[mt][r] + bias);
      }
  }
  __syncthreads();

  // ---- layer 3: [48,128] x [128,128], sM2 -> registers ----
  {
    acc3[0] = v8f{}; acc3[1] = v8f{}; acc3[2] = v8f{};
    const __bf16* sM2 = sA;
    for (int kt = 0; kt < HH / 32; ++kt) {
      v16bf bf = load_b(WT3, ncol, HH, kt * 32, hi);
#pragma unroll
      for (int mt = 0; mt < 3; ++mt) {
        v16bf af = load_a(sM2 + (mt * 16 + lo) * HH, kt * 32, hi);
        acc3[mt] = wmma_bf16(af, bf, acc3[mt]);
      }
    }
    *bias3_out = b3v[ncol];
  }
}

// ---------------- weight convert: W[k][128] f32 -> WT[n][kdim] bf16 ----------
__global__ void convert_wT(const float* __restrict__ W, __bf16* __restrict__ WT,
                           int kdim) {
  int i = blockIdx.x * 256 + threadIdx.x;
  int tot = kdim * 128;
  if (i < tot) {
    int n = i / kdim;
    int k = i - n * kdim;
    WT[i] = (__bf16)W[(size_t)k * 128 + n];
  }
}

// ---------------- node update + FFN -----------------------------------------
__global__ __launch_bounds__(256) void node_kernel(
    const float* __restrict__ hV, const float* __restrict__ hE,
    const int* __restrict__ Eidx, const float* __restrict__ maskV,
    const float* __restrict__ maskAtt,
    const __bf16* __restrict__ W1T, const float* __restrict__ b1,
    const __bf16* __restrict__ W2T, const float* __restrict__ b2,
    const __bf16* __restrict__ W3T, const float* __restrict__ b3,
    const float* __restrict__ Wi, const float* __restrict__ bi,
    const float* __restrict__ Wo, const float* __restrict__ bo,
    const float* __restrict__ ln1s, const float* __restrict__ ln1b,
    const float* __restrict__ ln2s, const float* __restrict__ ln2b,
    float* __restrict__ hVout) {
  __shared__ __bf16 sA[KN * IN3] __attribute__((aligned(16)));  // 36864 B
  __shared__ __bf16 sM[KN * HH] __attribute__((aligned(16)));   // 12288 B
  __shared__ float sRed[HH];
  __shared__ float sX[HH];
  __shared__ float sFF[FFD];
  __shared__ int sIdx[KN];
  __shared__ float sMask[KN];

  const int tid = threadIdx.x;
  const int bl = blockIdx.x;
  const int b = bl / LL;
  const size_t blH = (size_t)bl * HH;
  const size_t blK = (size_t)bl * KN;

  if (tid < KN) {
    sIdx[tid] = Eidx[blK + tid];
    sMask[tid] = maskAtt[blK + tid];
  }
  __syncthreads();

  // build h_EV = [center | h_E | neighbor] in bf16
  for (int t = tid; t < KN * IN3; t += 256) {
    int k = t / IN3, c = t - k * IN3;
    float v;
    if (c < HH)            v = hV[blH + c];
    else if (c < 2 * HH)   v = hE[(blK + k) * HH + (c - HH)];
    else                   v = hV[((size_t)b * LL + sIdx[k]) * HH + (c - 2 * HH)];
    sA[t] = (__bf16)v;
  }
  __syncthreads();

  v8f acc3[3];
  float bias3;
  mlp3_accum(sA, sM, W1T, b1, W2T, b2, W3T, b3, tid, acc3, &bias3);

  // layer3 epilogue: mask_attend * (acc + b3) -> sO f32 (aliases sA high part)
  float* sO = (float*)(sA + KN * HH);  // byte offset 12288, 24576 B
  {
    const int w = tid >> 5, lane = tid & 31, hi = lane >> 4, lo = lane & 15;
    const int ncol = w * 16 + lo;
#pragma unroll
    for (int mt = 0; mt < 3; ++mt)
#pragma unroll
      for (int r = 0; r < 8; ++r) {
        int row = mt * 16 + hi * 8 + r;
        sO[row * HH + ncol] = (acc3[mt][r] + bias3) * sMask[row];
      }
  }
  __syncthreads();

  // sum over K, residual, LN1
  float x = 0.0f;
  if (tid < HH) {
    float s = 0.0f;
    for (int k = 0; k < KN; ++k) s += sO[k * HH + tid];
    x = hV[blH + tid] + s * INV_SCALE;
    sRed[tid] = x;
  }
  float mu = blocksum128(sRed, tid) * (1.0f / HH);
  if (tid < HH) { float d = x - mu; sRed[tid] = d * d; }
  float var = blocksum128(sRed, tid) * (1.0f / HH);
  float rstd = rsqrtf(var + LN_EPS);
  if (tid < HH) sX[tid] = (x - mu) * rstd * ln1s[tid] + ln1b[tid];
  __syncthreads();

  // FFN: 128 -> 512 (gelu) -> 128
  {
    float a0 = bi[tid], a1 = bi[tid + 256];
    for (int f = 0; f < HH; ++f) {
      float xv = sX[f];
      a0 += xv * Wi[f * FFD + tid];
      a1 += xv * Wi[f * FFD + tid + 256];
    }
    sFF[tid] = gelu_erf(a0);
    sFF[tid + 256] = gelu_erf(a1);
  }
  __syncthreads();
  float x2 = 0.0f;
  if (tid < HH) {
    float o = bo[tid];
    for (int n = 0; n < FFD; ++n) o += sFF[n] * Wo[n * HH + tid];
    x2 = sX[tid] + o;
    sRed[tid] = x2;
  }
  float mu2 = blocksum128(sRed, tid) * (1.0f / HH);
  if (tid < HH) { float d = x2 - mu2; sRed[tid] = d * d; }
  float var2 = blocksum128(sRed, tid) * (1.0f / HH);
  float rstd2 = rsqrtf(var2 + LN_EPS);
  if (tid < HH)
    hVout[blH + tid] = ((x2 - mu2) * rstd2 * ln2s[tid] + ln2b[tid]) * maskV[bl];
}

// ---------------- edge update ------------------------------------------------
__global__ __launch_bounds__(256) void edge_kernel(
    const float* __restrict__ hVnew, const float* __restrict__ hE,
    const int* __restrict__ Eidx,
    const __bf16* __restrict__ W11T, const float* __restrict__ b11,
    const __bf16* __restrict__ W12T, const float* __restrict__ b12,
    const __bf16* __restrict__ W13T, const float* __restrict__ b13,
    const float* __restrict__ ln3s, const float* __restrict__ ln3b,
    float* __restrict__ hEout) {
  __shared__ __bf16 sA[KN * IN3] __attribute__((aligned(16)));
  __shared__ __bf16 sM[KN * HH] __attribute__((aligned(16)));
  __shared__ int sIdx[KN];
  __shared__ float sMu[KN];
  __shared__ float sRstd[KN];

  const int tid = threadIdx.x;
  const int bl = blockIdx.x;
  const int b = bl / LL;
  const size_t blH = (size_t)bl * HH;
  const size_t blK = (size_t)bl * KN;

  if (tid < KN) sIdx[tid] = Eidx[blK + tid];
  __syncthreads();

  for (int t = tid; t < KN * IN3; t += 256) {
    int k = t / IN3, c = t - k * IN3;
    float v;
    if (c < HH)            v = hVnew[blH + c];
    else if (c < 2 * HH)   v = hE[(blK + k) * HH + (c - HH)];
    else                   v = hVnew[((size_t)b * LL + sIdx[k]) * HH + (c - 2 * HH)];
    sA[t] = (__bf16)v;
  }
  __syncthreads();

  v8f acc3[3];
  float bias3;
  mlp3_accum(sA, sM, W11T, b11, W12T, b12, W13T, b13, tid, acc3, &bias3);

  // epilogue: residual h_E + m -> sO f32
  float* sO = (float*)(sA + KN * HH);
  {
    const int w = tid >> 5, lane = tid & 31, hi = lane >> 4, lo = lane & 15;
    const int ncol = w * 16 + lo;
#pragma unroll
    for (int mt = 0; mt < 3; ++mt)
#pragma unroll
      for (int r = 0; r < 8; ++r) {
        int row = mt * 16 + hi * 8 + r;
        sO[row * HH + ncol] =
            acc3[mt][r] + bias3 + hE[(blK + row) * HH + ncol];
      }
  }
  __syncthreads();

  // per-row LN3
  if (tid < KN) {
    const float* rowp = sO + tid * HH;
    float s = 0.0f;
    for (int f = 0; f < HH; ++f) s += rowp[f];
    float mu = s * (1.0f / HH);
    float v = 0.0f;
    for (int f = 0; f < HH; ++f) { float d = rowp[f] - mu; v += d * d; }
    sMu[tid] = mu;
    sRstd[tid] = rsqrtf(v * (1.0f / HH) + LN_EPS);
  }
  __syncthreads();
  for (int t = tid; t < KN * HH; t += 256) {
    int row = t >> 7, f = t & (HH - 1);
    hEout[blK * HH + t] = (sO[t] - sMu[row]) * sRstd[row] * ln3s[f] + ln3b[f];
  }
}

extern "C" void kernel_launch(void* const* d_in, const int* in_sizes, int n_in,
                              void* d_out, int out_size, void* d_ws, size_t ws_size,
                              hipStream_t stream) {
  const float* hV = (const float*)d_in[0];
  const float* hE = (const float*)d_in[1];
  const int* Eidx = (const int*)d_in[2];
  const float* maskV = (const float*)d_in[3];
  const float* maskAtt = (const float*)d_in[4];
  const float* W1 = (const float*)d_in[5];   const float* b1 = (const float*)d_in[6];
  const float* W2 = (const float*)d_in[7];   const float* b2 = (const float*)d_in[8];
  const float* W3 = (const float*)d_in[9];   const float* b3 = (const float*)d_in[10];
  const float* W11 = (const float*)d_in[11]; const float* b11 = (const float*)d_in[12];
  const float* W12 = (const float*)d_in[13]; const float* b12 = (const float*)d_in[14];
  const float* W13 = (const float*)d_in[15]; const float* b13 = (const float*)d_in[16];
  const float* Wi = (const float*)d_in[17];  const float* bi = (const float*)d_in[18];
  const float* Wo = (const float*)d_in[19];  const float* bo = (const float*)d_in[20];
  const float* ln1s = (const float*)d_in[21]; const float* ln1b = (const float*)d_in[22];
  const float* ln2s = (const float*)d_in[23]; const float* ln2b = (const float*)d_in[24];
  const float* ln3s = (const float*)d_in[25]; const float* ln3b = (const float*)d_in[26];

  // bf16 transposed weights in workspace: total 163840 bf16 = 320 KB
  __bf16* W1T = (__bf16*)d_ws;
  __bf16* W2T = W1T + 128 * 384;
  __bf16* W3T = W2T + 128 * 128;
  __bf16* W11T = W3T + 128 * 128;
  __bf16* W12T = W11T + 128 * 384;
  __bf16* W13T = W12T + 128 * 128;

  convert_wT<<<(128 * 384 + 255) / 256, 256, 0, stream>>>(W1, W1T, 384);
  convert_wT<<<(128 * 128 + 255) / 256, 256, 0, stream>>>(W2, W2T, 128);
  convert_wT<<<(128 * 128 + 255) / 256, 256, 0, stream>>>(W3, W3T, 128);
  convert_wT<<<(128 * 384 + 255) / 256, 256, 0, stream>>>(W11, W11T, 384);
  convert_wT<<<(128 * 128 + 255) / 256, 256, 0, stream>>>(W12, W12T, 128);
  convert_wT<<<(128 * 128 + 255) / 256, 256, 0, stream>>>(W13, W13T, 128);

  float* hVout = (float*)d_out;
  float* hEout = hVout + (size_t)NBLK * HH;

  node_kernel<<<NBLK, 256, 0, stream>>>(hV, hE, Eidx, maskV, maskAtt,
                                        W1T, b1, W2T, b2, W3T, b3,
                                        Wi, bi, Wo, bo,
                                        ln1s, ln1b, ln2s, ln2b, hVout);
  edge_kernel<<<NBLK, 256, 0, stream>>>(hVout, hE, Eidx,
                                        W11T, b11, W12T, b12, W13T, b13,
                                        ln3s, ln3b, hEout);
}
